// MoELayer_22677427323477
// MI455X (gfx1250) — matrix operand order repaired and checked by reading, not compiled, not run on hardware
//
#include <hip/hip_runtime.h>

// ---------------------------------------------------------------------------
// MoE layer for MI455X (gfx1250): fp16 WMMA GEMMs (v_wmma_f32_16x16x32_f16),
// fp32 accumulate, fused bias/activation/residual epilogues.
// 64x128 tile / workgroup (8 wave32s), 4 WMMAs per wave per 32-K step.
// ---------------------------------------------------------------------------

typedef __attribute__((ext_vector_type(16))) _Float16 v16h;
typedef __attribute__((ext_vector_type(8)))  float    v8f;

#define TOKS  4096   // B*S
#define HDIM  2048   // H
#define HRDIM 1024   // H/2 (router hidden)
#define NEXP  3      // E
#define FDIM  8192   // 4*H

union Frag16 { v16h v; unsigned int u[8]; };

__device__ __forceinline__ void pack_store_f16x2(_Float16* dst, float lo, float hi)
{
    union { _Float16 h[2]; unsigned int u; } pk;
    pk.h[0] = (_Float16)lo;
    pk.h[1] = (_Float16)hi;
    *(unsigned int*)dst = pk.u;   // dst is 4B-aligned (k even, row pitch 40 f16)
}

// ---------------------------------------------------------------------------
// fp32 -> fp16 conversion (activations)
// ---------------------------------------------------------------------------
__global__ __launch_bounds__(256)
void cvt_f32_to_f16_kernel(const float* __restrict__ src,
                           _Float16* __restrict__ dst, int n)
{
    int i = (blockIdx.x * 256 + threadIdx.x) * 4;
    if (i + 3 < n) {
        float4 f = *(const float4*)(src + i);
        dst[i + 0] = (_Float16)f.x;
        dst[i + 1] = (_Float16)f.y;
        dst[i + 2] = (_Float16)f.z;
        dst[i + 3] = (_Float16)f.w;
    }
}

// ---------------------------------------------------------------------------
// Tiled WMMA GEMM:  C[M,N] = act(A[M,K] @ Bw[K,N] + bias[N]) (+epilogue)
//   A  : f16, row-major, stride K (pre-converted in workspace)
//   Bw : f32, row-major, stride N (weights from global; cvt while staging)
// ACT: 0=none 1=relu 2=gelu(exact)   OUTMODE: 0=f16 store  1=expert-select f32
// ---------------------------------------------------------------------------
template<int ACT, int OUTMODE>
__global__ __launch_bounds__(256)
void gemm_wmma_kernel(const _Float16* __restrict__ A, int K,
                      const float* __restrict__ Bw, int N,
                      const float* __restrict__ bias,
                      _Float16* __restrict__ out16,
                      float* __restrict__ outf,
                      const float* __restrict__ xres,
                      const int* __restrict__ eidx,
                      int expert)
{
    // stride-40 (=20 dword) row pitch: conflict-free frag reads, aligned stores
    __shared__ _Float16 AL[64][40];    // A tile  [m][k]   64x32
    __shared__ _Float16 BL[128][40];   // B tile  [n][k]   (transposed) 128x32

    const int tid  = threadIdx.x;
    const int lane = tid & 31;
    const int wave = tid >> 5;
    const int half = lane >> 4;        // 0: lanes 0-15, 1: lanes 16-31
    const int lm   = lane & 15;
    const int tile_m = blockIdx.y * 64;
    const int tile_n = blockIdx.x * 128;
    const int wm = (wave & 3) * 16;    // wave's M sub-block
    const int wn = (wave >> 2) * 64;   // wave's 4-wide N sub-block base

    // cooperative-load indices
    const int ar = tid >> 2;           // 0..63  A-tile row
    const int ak = (tid & 3) * 8;      // 0,8,16,24
    const int bk = (tid >> 4) * 2;     // 0,2,..,30  B-tile even k-row
    const int bc = (tid & 15) * 8;     // 0..120     B-tile col base

    v8f acc[4];
#pragma unroll
    for (int b = 0; b < 4; ++b)
        acc[b] = (v8f){0.f, 0.f, 0.f, 0.f, 0.f, 0.f, 0.f, 0.f};

    for (int k0 = 0; k0 < K; k0 += 32) {
        __syncthreads();
        // --- stage A tile: 64x32 f16 (one uint4 per thread) ---
        {
            const _Float16* s = A + (size_t)(tile_m + ar) * K + (k0 + ak);
            *(uint4*)&AL[ar][ak] = *(const uint4*)s;
        }
        // --- stage B tile: 32x128 f32 -> f16, transposed+packed into BL ---
        {
            const float* s = Bw + (size_t)(k0 + bk) * N + (tile_n + bc);
            float4 a0 = *(const float4*)s;
            float4 a1 = *(const float4*)(s + 4);
            float4 c0 = *(const float4*)(s + N);
            float4 c1 = *(const float4*)(s + N + 4);
            pack_store_f16x2(&BL[bc + 0][bk], a0.x, c0.x);
            pack_store_f16x2(&BL[bc + 1][bk], a0.y, c0.y);
            pack_store_f16x2(&BL[bc + 2][bk], a0.z, c0.z);
            pack_store_f16x2(&BL[bc + 3][bk], a0.w, c0.w);
            pack_store_f16x2(&BL[bc + 4][bk], a1.x, c1.x);
            pack_store_f16x2(&BL[bc + 5][bk], a1.y, c1.y);
            pack_store_f16x2(&BL[bc + 6][bk], a1.z, c1.z);
            pack_store_f16x2(&BL[bc + 7][bk], a1.w, c1.w);
            if (k0 + 32 < K)  // global_prefetch_b8 for next B tile
                __builtin_prefetch(s + (size_t)32 * N, 0, 1);
        }
        __syncthreads();

        // --- A fragment (ISA 16x32 f16 layout: lanes0-15 K {0..7,16..23}) ---
        Frag16 af;
#pragma unroll
        for (int p = 0; p < 8; ++p) {
            const int ka = ((p < 4) ? (2 * p) : (2 * p + 8)) + 8 * half;
            af.u[p] = *(const unsigned int*)&AL[wm + lm][ka];
        }
        // --- 4 B fragments, 4 WMMAs, A reused ---
#pragma unroll
        for (int b = 0; b < 4; ++b) {
            Frag16 bf;
#pragma unroll
            for (int p = 0; p < 8; ++p) {
                const int kb = ((p < 4) ? (2 * p) : (2 * p + 8)) + 8 * half;   // lane = column, K {0..7,16..23} / {8..15,24..31}
                bf.u[p] = *(const unsigned int*)&BL[wn + 16 * b + lm][kb];
            }
            acc[b] = __builtin_amdgcn_wmma_f32_16x16x32_f16(
                         false, af.v, false, bf.v, (short)0, acc[b], false, false);
        }
    }

    // --- epilogue: C/D layout is VGPR r -> M = r + 8*half, N = lane&15 ---
#pragma unroll
    for (int r = 0; r < 8; ++r) {
        const int row = tile_m + wm + 8 * half + r;
#pragma unroll
        for (int b = 0; b < 4; ++b) {
            const int col = tile_n + wn + 16 * b + lm;
            float v = acc[b][r];
            v += bias[col];
            if (ACT == 1) v = fmaxf(v, 0.f);
            else if (ACT == 2) v = 0.5f * v * (1.0f + erff(v * 0.70710678118654752f));
            if (OUTMODE == 0) {
                out16[(size_t)row * N + col] = (_Float16)v;
            } else {
                if (eidx[row] == expert)  // exactly one expert matches per row
                    outf[(size_t)row * N + col] = v + xres[(size_t)row * N + col];
            }
        }
    }
}

// ---------------------------------------------------------------------------
// Router second layer (N=3, too thin for WMMA): per-token dot products,
// logits to d_out tail, first-max argmax -> eidx.
// ---------------------------------------------------------------------------
__global__ __launch_bounds__(128)
void router_logits_kernel(const _Float16* __restrict__ midr,
                          const float* __restrict__ wr2,
                          const float* __restrict__ br2,
                          float* __restrict__ logits,
                          int* __restrict__ eidx)
{
    __shared__ float red[NEXP][128];
    const int tok = blockIdx.x;
    const int tid = threadIdx.x;
    float s0 = 0.f, s1 = 0.f, s2 = 0.f;
    for (int j = tid; j < HRDIM; j += 128) {
        float m = (float)midr[(size_t)tok * HRDIM + j];
        s0 += m * wr2[j * NEXP + 0];
        s1 += m * wr2[j * NEXP + 1];
        s2 += m * wr2[j * NEXP + 2];
    }
    red[0][tid] = s0; red[1][tid] = s1; red[2][tid] = s2;
    __syncthreads();
    for (int off = 64; off > 0; off >>= 1) {
        if (tid < off) {
            red[0][tid] += red[0][tid + off];
            red[1][tid] += red[1][tid + off];
            red[2][tid] += red[2][tid + off];
        }
        __syncthreads();
    }
    if (tid == 0) {
        float l0 = red[0][0] + br2[0];
        float l1 = red[1][0] + br2[1];
        float l2 = red[2][0] + br2[2];
        logits[tok * NEXP + 0] = l0;
        logits[tok * NEXP + 1] = l1;
        logits[tok * NEXP + 2] = l2;
        int best = 0; float bv = l0;
        if (l1 > bv) { bv = l1; best = 1; }   // strict >: first-max like argmax
        if (l2 > bv) { bv = l2; best = 2; }
        eidx[tok] = best;
    }
}

// ---------------------------------------------------------------------------
extern "C" void kernel_launch(void* const* d_in, const int* in_sizes, int n_in,
                              void* d_out, int out_size, void* d_ws, size_t ws_size,
                              hipStream_t stream)
{
    (void)in_sizes; (void)n_in; (void)out_size; (void)ws_size;

    const float* x   = (const float*)d_in[0];   // [TOKS, H]
    const float* wr1 = (const float*)d_in[1];   // [H, H/2]
    const float* br1 = (const float*)d_in[2];   // [H/2]
    const float* wr2 = (const float*)d_in[3];   // [H/2, E]
    const float* br2 = (const float*)d_in[4];   // [E]
    const float* w1  = (const float*)d_in[5];   // [E, H, F]
    const float* b1  = (const float*)d_in[6];   // [E, F]
    const float* w2  = (const float*)d_in[7];   // [E, F, H]
    const float* b2  = (const float*)d_in[8];   // [E, H]

    float* out    = (float*)d_out;                       // [TOKS, H]
    float* logits = out + (size_t)TOKS * HDIM;           // [TOKS, E]

    // workspace layout
    char* ws = (char*)d_ws;
    _Float16* xh   = (_Float16*)ws;                                       // 16 MB
    _Float16* midr = (_Float16*)(ws + (size_t)TOKS * HDIM * 2);           //  8 MB
    _Float16* midm = (_Float16*)(ws + (size_t)TOKS * (HDIM + HRDIM) * 2); // 64 MB
    int*      eidx = (int*)(ws + (size_t)TOKS * (HDIM + HRDIM + FDIM) * 2);

    // 1) x -> f16
    cvt_f32_to_f16_kernel<<<(TOKS * HDIM) / 1024, 256, 0, stream>>>(
        x, xh, TOKS * HDIM);

    // 2) router GEMM1: midr = relu(x @ wr1 + br1)     [4096,1024]
    gemm_wmma_kernel<1, 0><<<dim3(HRDIM / 128, TOKS / 64), 256, 0, stream>>>(
        xh, HDIM, wr1, HRDIM, br1, midr, nullptr, nullptr, nullptr, 0);

    // 3) router logits + argmax
    router_logits_kernel<<<TOKS, 128, 0, stream>>>(midr, wr2, br2, logits, eidx);

    // 4) experts: dense GEMM1(gelu) then GEMM2 with expert-select + residual
    for (int e = 0; e < NEXP; ++e) {
        gemm_wmma_kernel<2, 0><<<dim3(FDIM / 128, TOKS / 64), 256, 0, stream>>>(
            xh, HDIM, w1 + (size_t)e * HDIM * FDIM, FDIM,
            b1 + (size_t)e * FDIM, midm, nullptr, nullptr, nullptr, 0);

        gemm_wmma_kernel<0, 1><<<dim3(HDIM / 128, TOKS / 64), 256, 0, stream>>>(
            midm, FDIM, w2 + (size_t)e * FDIM * HDIM, HDIM,
            b2 + (size_t)e * HDIM, nullptr, out, x, eidx, e);
    }
}
